// LIIF_10376640987217
// MI455X (gfx1250) — compile-verified, matrix operand order checked by reading, not run on hardware
//
#include <hip/hip_runtime.h>

typedef __attribute__((ext_vector_type(16))) _Float16 v16h;
typedef __attribute__((ext_vector_type(8)))  float    v8f;

#define BATCH 2
#define CCH 64
#define HH 64
#define WW 64
#define QQ 65536
#define HIDN 256
#define IN_DIM 580
#define IN_PAD 608
#define KT0 19           // 608/32 K-tiles for layer 0
#define KTH 8            // 256/32 K-tiles for hidden layers
#define NTH 16           // 256/16 N-tiles
#define MT  4            // M-tiles per workgroup (64 rows)
#define ROWS_TOTAL (BATCH*QQ*4)        // 524288
#define ROWS_PER_WG (MT*16)            // 64
#define NWG (ROWS_TOTAL/ROWS_PER_WG)   // 8192

// ---- workspace layout (bytes) ----
#define OFF_W0 0
#define SZ_W0  (KT0*NTH*512*2)          // 311296
#define OFF_W1 (OFF_W0 + SZ_W0)
#define SZ_WH  (KTH*NTH*512*2)          // 131072
#define OFF_W2 (OFF_W1 + SZ_WH)
#define OFF_W3 (OFF_W2 + SZ_WH)
#define OFF_W4 (OFF_W3 + SZ_WH)
#define SZ_W4  (KTH*1*512*2)            // 8192
#define OFF_PRED (OFF_W4 + SZ_W4)
#define SZ_PRED (ROWS_TOTAL*3*4)        // 6291456
#define OFF_AREA (OFF_PRED + SZ_PRED)

// ---------------------------------------------------------------------------
// Weight preconversion: f32 row-major [Kreal x Nreal] -> f16 WMMA B-operand
// tiles [kt][nt][lane(32)][half(16)], zero-padded to KT*32 x NT*16.
//   lane = (n%16) + 16*((kk>>3)&1), half = ((kk>>4)<<3) | (kk&7), kk = k%32
// ---------------------------------------------------------------------------
__global__ void prep_w_kernel(const float* __restrict__ w, _Float16* __restrict__ dst,
                              int Kreal, int Nreal, int KT, int NT) {
  int total = KT * NT * 512;
  for (int e = blockIdx.x * blockDim.x + threadIdx.x; e < total;
       e += gridDim.x * blockDim.x) {
    int tile = e >> 9;
    int r    = e & 511;
    int lane = r >> 4;
    int half = r & 15;
    int nt = tile % NT;
    int kt = tile / NT;
    int kk = ((half >> 3) << 4) | ((lane >> 4) << 3) | (half & 7);
    int k  = kt * 32 + kk;
    int n  = nt * 16 + (lane & 15);
    float v = (k < Kreal && n < Nreal) ? w[(size_t)k * Nreal + n] : 0.0f;
    dst[e] = (_Float16)v;
  }
}

// ---------------------------------------------------------------------------
__device__ __forceinline__ v8f wmma16(v16h a, v16h b, v8f c) {
  return __builtin_amdgcn_wmma_f32_16x16x32_f16(false, a, false, b, (short)0, c,
                                                false, false);
}

// store one f32 activation element (M=mt*16+mrow, N=ntX*16+nlow) into the
// next layer's LDS A-operand tiles (MT M-tiles), as f16.
__device__ __forceinline__ void storeAct(_Float16* dst, int mt, int ntX,
                                         int mrow, int nlow, float y) {
  int kt2   = ntX >> 1;
  int lane2 = mrow + (((nlow >> 3) & 1) << 4);
  int half2 = ((ntX & 1) << 3) | (nlow & 7);
  dst[(kt2 * MT + mt) * 512 + lane2 * 16 + half2] = (_Float16)y;
}

// One dense layer: [64 x 32*KT] (LDS, A-layout) x [32*KT x 256] -> [64 x 256]
// Each of 8 waves owns N-tiles {2*wv, 2*wv+1} and all MT M-tiles.
template <int KT>
__device__ __forceinline__ void layer_hidden(const _Float16* __restrict__ src,
                                             _Float16* __restrict__ dst,
                                             const _Float16* __restrict__ Wt,
                                             const float* __restrict__ bias,
                                             int wv, int lane, bool relu) {
  const int nt0 = 2 * wv, nt1 = nt0 + 1;
  v8f acc[MT][2];
#pragma unroll
  for (int mt = 0; mt < MT; ++mt) {
    acc[mt][0] = (v8f){};
    acc[mt][1] = (v8f){};
  }
#pragma unroll
  for (int kt = 0; kt < KT; ++kt) {
    v16h b0 = *(const v16h*)(Wt + ((size_t)((kt * NTH + nt0) * 32 + lane)) * 16);
    v16h b1 = *(const v16h*)(Wt + ((size_t)((kt * NTH + nt1) * 32 + lane)) * 16);
#pragma unroll
    for (int mt = 0; mt < MT; ++mt) {
      v16h x = *(const v16h*)(src + (kt * MT + mt) * 512 + lane * 16);
      acc[mt][0] = wmma16(x, b0, acc[mt][0]);
      acc[mt][1] = wmma16(x, b1, acc[mt][1]);
    }
  }
  int nlow = lane & 15, hib = lane >> 4;
  float bi0 = bias[nt0 * 16 + nlow];
  float bi1 = bias[nt1 * 16 + nlow];
#pragma unroll
  for (int mt = 0; mt < MT; ++mt) {
#pragma unroll
    for (int v = 0; v < 8; ++v) {
      int mrow = v + 8 * hib;  // M % 16 within tile
      float y;
      y = acc[mt][0][v] + bi0; if (relu) y = fmaxf(y, 0.0f);
      storeAct(dst, mt, nt0, mrow, nlow, y);
      y = acc[mt][1][v] + bi1; if (relu) y = fmaxf(y, 0.0f);
      storeAct(dst, mt, nt1, mrow, nlow, y);
    }
  }
}

// ---------------------------------------------------------------------------
// Main kernel: 64 rows (16 queries x 4 corners) per workgroup, 5-layer MLP.
// ---------------------------------------------------------------------------
__global__ __launch_bounds__(256) void liif_mlp_kernel(
    const float* __restrict__ feat, const float* __restrict__ coord,
    const float* __restrict__ cell, const float* __restrict__ b0,
    const float* __restrict__ b1, const float* __restrict__ b2,
    const float* __restrict__ b3, const float* __restrict__ b4,
    const _Float16* __restrict__ W0t, const _Float16* __restrict__ W1t,
    const _Float16* __restrict__ W2t, const _Float16* __restrict__ W3t,
    const _Float16* __restrict__ W4t, float* __restrict__ preds,
    float* __restrict__ areas) {
  __shared__ __align__(64) _Float16 ldsA[KT0 * MT * 512];  // 77824 B
  __shared__ __align__(64) _Float16 ldsB[KTH * MT * 512];  // 32768 B
  __shared__ int   s_iy[ROWS_PER_WG], s_ix[ROWS_PER_WG];
  __shared__ float s_ext[ROWS_PER_WG][4];  // cs0, cs1, rel0, rel1

  const int tid = threadIdx.x;
  const int rowBase = blockIdx.x * ROWS_PER_WG;
  const int b = rowBase / (QQ * 4);  // uniform across the block

  // ---- phase 1: per-row metadata (grid-sample nearest, rel, cell, area) ----
  if (tid < ROWS_PER_WG) {
    int r = rowBase + tid;
    int rem = r - b * (QQ * 4);
    int q = rem >> 2;
    int corner = rem & 3;                       // vx = bit1, vy = bit0
    float vx = (corner & 2) ? 1.0f : -1.0f;
    float vy = (corner & 1) ? 1.0f : -1.0f;
    const float* cp = coord + ((size_t)b * QQ + q) * 2;
    const float* ce = cell + ((size_t)b * QQ + q) * 2;
    float c0 = cp[0], c1 = cp[1];
    const float eps = 1e-6f;
    float s0 = fminf(fmaxf(c0 + vx * (1.0f / HH) + eps, -1.0f + 1e-6f), 1.0f - 1e-6f);
    float s1 = fminf(fmaxf(c1 + vy * (1.0f / WW) + eps, -1.0f + 1e-6f), 1.0f - 1e-6f);
    int iy = (int)rintf((s0 + 1.0f) * (HH * 0.5f) - 0.5f);
    iy = min(max(iy, 0), HH - 1);
    int ix = (int)rintf((s1 + 1.0f) * (WW * 0.5f) - 0.5f);
    ix = min(max(ix, 0), WW - 1);
    float fy = ((float)iy + 0.5f) * (2.0f / HH) - 1.0f;
    float fx = ((float)ix + 0.5f) * (2.0f / WW) - 1.0f;
    float rel0 = (c0 - fy) * (float)HH;
    float rel1 = (c1 - fx) * (float)WW;
    s_iy[tid] = iy;
    s_ix[tid] = ix;
    s_ext[tid][0] = ce[0] * (float)HH;
    s_ext[tid][1] = ce[1] * (float)WW;
    s_ext[tid][2] = rel0;
    s_ext[tid][3] = rel1;
    areas[r] = fabsf(rel0 * rel1) + 1e-9f;
  }
  __syncthreads();

  // ---- phase 2: gather/unfold -> LDS A0 in WMMA A-operand layout ----
  const float* fb = feat + (size_t)b * CCH * HH * WW;
  for (int e = tid; e < ROWS_PER_WG * IN_PAD; e += 256) {
    int row = e / IN_PAD;
    int f = e - row * IN_PAD;
    float v;
    if (f < 576) {
      int c = f / 9, tap = f - c * 9;
      int ki = tap / 3, kj = tap - ki * 3;
      int yy = s_iy[row] + ki - 1;
      int xx = s_ix[row] + kj - 1;
      v = (yy >= 0 && yy < HH && xx >= 0 && xx < WW)
              ? fb[((size_t)c * HH + yy) * WW + xx]
              : 0.0f;
    } else if (f < IN_DIM) {
      v = s_ext[row][f - 576];
    } else {
      v = 0.0f;  // K padding 580..607
    }
    int mt = row >> 4, m = row & 15;
    int kt = f >> 5, kk = f & 31;
    int lane = m + (((kk >> 3) & 1) << 4);
    int half = ((kk >> 4) << 3) | (kk & 7);
    ldsA[(kt * MT + mt) * 512 + lane * 16 + half] = (_Float16)v;
  }
  __syncthreads();

  // ---- phase 3: MLP via WMMA ----
  const int wv = tid >> 5;
  const int lane = tid & 31;

  layer_hidden<KT0>(ldsA, ldsB, W0t, b0, wv, lane, true);
  __syncthreads();
  layer_hidden<KTH>(ldsB, ldsA, W1t, b1, wv, lane, true);
  __syncthreads();
  layer_hidden<KTH>(ldsA, ldsB, W2t, b2, wv, lane, true);
  __syncthreads();
  layer_hidden<KTH>(ldsB, ldsA, W3t, b3, wv, lane, true);
  __syncthreads();

  // final layer 256 -> 3 (padded N=16): waves 0..3 take M-tiles 0..3
  if (wv < MT) {
    const int mt = wv;
    v8f acc = {};
#pragma unroll
    for (int kt = 0; kt < KTH; ++kt) {
      v16h bo = *(const v16h*)(W4t + ((size_t)(kt * 32 + lane)) * 16);
      v16h xo = *(const v16h*)(ldsA + (kt * MT + mt) * 512 + lane * 16);
      acc = wmma16(xo, bo, acc);
    }
    int nlow = lane & 15, hib = lane >> 4;
    if (nlow < 3) {
      float bi = b4[nlow];
#pragma unroll
      for (int v = 0; v < 8; ++v) {
        int M = mt * 16 + v + 8 * hib;
        preds[(size_t)(rowBase + M) * 3 + nlow] = acc[v] + bi;
      }
    }
  }
}

// ---------------------------------------------------------------------------
// Local-ensemble combine: ret = sum_c pred[c] * area[3-c] / tot
// ---------------------------------------------------------------------------
__global__ void combine_kernel(const float* __restrict__ preds,
                               const float* __restrict__ areas,
                               float* __restrict__ out) {
  int idx = blockIdx.x * 256 + threadIdx.x;
  if (idx >= BATCH * QQ) return;
  int base = idx * 4;
  float a0 = areas[base + 0], a1 = areas[base + 1];
  float a2 = areas[base + 2], a3 = areas[base + 3];
  float inv = 1.0f / (a0 + a1 + a2 + a3);
  float g0 = a3 * inv, g1 = a2 * inv, g2 = a1 * inv, g3 = a0 * inv;
#pragma unroll
  for (int n = 0; n < 3; ++n) {
    float s = preds[(size_t)(base + 0) * 3 + n] * g0 +
              preds[(size_t)(base + 1) * 3 + n] * g1 +
              preds[(size_t)(base + 2) * 3 + n] * g2 +
              preds[(size_t)(base + 3) * 3 + n] * g3;
    out[(size_t)idx * 3 + n] = s;
  }
}

// ---------------------------------------------------------------------------
extern "C" void kernel_launch(void* const* d_in, const int* in_sizes, int n_in,
                              void* d_out, int out_size, void* d_ws,
                              size_t ws_size, hipStream_t stream) {
  const float* feat  = (const float*)d_in[0];
  const float* coord = (const float*)d_in[1];
  const float* cell  = (const float*)d_in[2];
  const float* w0 = (const float*)d_in[3];
  const float* b0 = (const float*)d_in[4];
  const float* w1 = (const float*)d_in[5];
  const float* b1 = (const float*)d_in[6];
  const float* w2 = (const float*)d_in[7];
  const float* b2 = (const float*)d_in[8];
  const float* w3 = (const float*)d_in[9];
  const float* b3 = (const float*)d_in[10];
  const float* w4 = (const float*)d_in[11];
  const float* b4 = (const float*)d_in[12];

  char* ws = (char*)d_ws;
  _Float16* W0t = (_Float16*)(ws + OFF_W0);
  _Float16* W1t = (_Float16*)(ws + OFF_W1);
  _Float16* W2t = (_Float16*)(ws + OFF_W2);
  _Float16* W3t = (_Float16*)(ws + OFF_W3);
  _Float16* W4t = (_Float16*)(ws + OFF_W4);
  float* preds = (float*)(ws + OFF_PRED);
  float* areas = (float*)(ws + OFF_AREA);

  // weight f32 -> f16 WMMA-tile conversion (every call; deterministic)
  prep_w_kernel<<<(KT0 * NTH * 512 + 255) / 256, 256, 0, stream>>>(
      w0, W0t, IN_DIM, HIDN, KT0, NTH);
  prep_w_kernel<<<(KTH * NTH * 512 + 255) / 256, 256, 0, stream>>>(
      w1, W1t, HIDN, HIDN, KTH, NTH);
  prep_w_kernel<<<(KTH * NTH * 512 + 255) / 256, 256, 0, stream>>>(
      w2, W2t, HIDN, HIDN, KTH, NTH);
  prep_w_kernel<<<(KTH * NTH * 512 + 255) / 256, 256, 0, stream>>>(
      w3, W3t, HIDN, HIDN, KTH, NTH);
  prep_w_kernel<<<(KTH * 1 * 512 + 255) / 256, 256, 0, stream>>>(
      w4, W4t, HIDN, 3, KTH, 1);

  liif_mlp_kernel<<<NWG, 256, 0, stream>>>(feat, coord, cell, b0, b1, b2, b3,
                                           b4, W0t, W1t, W2t, W3t, W4t, preds,
                                           areas);

  combine_kernel<<<(BATCH * QQ + 255) / 256, 256, 0, stream>>>(preds, areas,
                                                               (float*)d_out);
}
